// ScaledDotProductAttention_9165460209763
// MI455X (gfx1250) — matrix operand order repaired and checked by reading, not compiled, not run on hardware
//
#include <hip/hip_runtime.h>
#include <math.h>
#include <stdint.h>

// ---------------------------------------------------------------------------
// Scaled dot-product attention for gfx1250 (MI455X), wave32 + WMMA + TDM.
//   scores = (Q/T) @ K^T ; attn = softmax(scores) ; out = attn @ V
// Shapes: [B=2, H=8, S=2048, D=64] fp32. d_out = [output | attn] concatenated.
//
// Roofline: attn write (268 MB) dominates -> ~13 us floor @ 23.3 TB/s.
// One workgroup (8 wave32) owns one (b,h) x 32-query block; the full
// 32x2048 f32 score block lives in LDS (256 KB of 320 KB/WGP) so attn
// bytes touch HBM exactly once. GEMMs use v_wmma_f32_16x16x32_f16.
// V tiles are DMA'd into double-buffered LDS by the Tensor Data Mover
// (tensor_load_to_lds + s_wait_tensorcnt), overlapping DMA with WMMA.
// ---------------------------------------------------------------------------

typedef __attribute__((ext_vector_type(16))) _Float16 v16h;
typedef __attribute__((ext_vector_type(8)))  float    v8f;
typedef unsigned int v4u __attribute__((ext_vector_type(4)));
typedef int          v8i __attribute__((ext_vector_type(8)));
typedef int          v4i __attribute__((ext_vector_type(4)));

#define SEQ   2048
#define DIM   64
#define BM    32            // queries per workgroup
#define T_INV 0.125f        // 1 / TEMPERATURE (T = 8 = sqrt(64))

#if __has_builtin(__builtin_amdgcn_tensor_load_to_lds)
#define USE_TDM 1
#endif

#ifdef USE_TDM
// Issue a TDM load of a 2D f32 tile [BM rows x DIM cols] starting at gsrc
// (row stride DIM) into LDS at lds_addr. D# layout per CDNA5 ISA ch.8.
// This toolchain exposes the 6-arg builtin:
//   (uint32x4 g0, int32x8 g1, int32x4 g2, int32x4 g3, int32x8 extra, i32 cpol)
__device__ __forceinline__ void tdm_issue_v_tile(const float* gsrc, uint32_t lds_addr)
{
    const uint64_t ga = (uint64_t)(uintptr_t)gsrc;
    v4u g0;
    g0[0] = 1u;                                           // count=1 (valid user D#)
    g0[1] = lds_addr;                                     // lds_addr[31:0]
    g0[2] = (uint32_t)ga;                                 // global_addr[31:0]
    g0[3] = (uint32_t)((ga >> 32) & 0x01FFFFFFu)          // global_addr[56:32]
          | (2u << 30);                                   // type = 2 ("image")
    v8i g1;
    g1[0] = 0x20000;                                      // data_size=2 (4 bytes)
    g1[1] = (int)((unsigned)DIM << 16);                   // tensor_dim0 = 64
    g1[2] = (int)((unsigned)SEQ << 16);                   // tensor_dim1 = 2048
    g1[3] = (int)((unsigned)DIM << 16);                   // tile_dim0 = 64
    g1[4] = BM;                                           // tile_dim1 = 32
    g1[5] = DIM;                                          // tensor_dim0_stride = 64
    g1[6] = 0;
    g1[7] = 0;
    const v4i z4 = {0, 0, 0, 0};                          // groups 2/3 unused (2D)
    const v8i z8 = {0, 0, 0, 0, 0, 0, 0, 0};              // unused trailing group
    __builtin_amdgcn_tensor_load_to_lds(g0, g1, z4, z4, z8, 0); // cpol = 0
}
#endif

__launch_bounds__(256, 1)
__global__ void sdpa_wmma_kernel(const float* __restrict__ q,
                                 const float* __restrict__ k,
                                 const float* __restrict__ v,
                                 float* __restrict__ out,
                                 float* __restrict__ attn)
{
    // ---- LDS: 262144 + 4096 + 16384 = 282624 B (< 320 KB/WGP) ----
    __shared__ float    sS[BM][SEQ];        // scores -> probabilities (f32)
    __shared__ _Float16 sQ[BM][DIM];        // Q block, pre-scaled, f16
    __shared__ float    sV[2][BM][DIM];     // double-buffered V k-tiles (f32)

    const int bh   = blockIdx.y;               // 0..15 (b*8+h)
    const int m0   = blockIdx.x * BM;          // query-block origin
    const int tid  = threadIdx.x;
    const int w    = tid >> 5;                 // wave id 0..7
    const int lane = tid & 31;
    const int lh   = lane >> 4;                // lane-half 0/1
    const int ll   = lane & 15;                // lane-in-half

    const size_t head = (size_t)bh * SEQ * DIM;
    const float* Q = q + head;
    const float* K = k + head;
    const float* V = v + head;

    // ---- stage Q block (scaled by 1/T) into LDS as f16 ----
    {
        const int r = tid >> 3, c = (tid & 7) * 8;
        const float* qp = Q + (size_t)(m0 + r) * DIM + c;
        #pragma unroll
        for (int i = 0; i < 8; ++i) sQ[r][c + i] = (_Float16)(qp[i] * T_INV);
    }
    __syncthreads();

    // ======================= Phase 1: S = (Q/T) @ K^T =======================
    // wave w: M-subtile = w>>2 (rows 16*(w>>2)..+15); key tiles (w&3)+4j.
    const int m_sub = w >> 2;
    const int Mrow  = m_sub * 16 + ll;

    // A-fragments (16x32 f16): a[e]=Q[M][lh*16+e] (+32)
    v16h a0, a1;
    {
        const _Float16* qrow = &sQ[Mrow][0];
        const int d0 = lh * 16;
        #pragma unroll
        for (int e = 0; e < 16; ++e) {
            a0[e]     = qrow[d0 + e];
            a1[e]     = qrow[32 + d0 + e];
        }
    }

    for (int j = 0; j < 32; ++j) {
        const int n0 = (w & 3) * 16 + j * 64;   // key-tile origin
        // B-fragment (32x16): B[kk][n] = K[n][kk]; lane: n=n0+ll, kk=lh*16+e.
        const float* krow = K + (size_t)(n0 + ll) * DIM + lh * 16;
        if (j + 1 < 32)                          // prefetch next key tile
            __builtin_prefetch(krow + 64 * DIM, 0, 0);
        v16h b0, b1;
        #pragma unroll
        for (int e = 0; e < 16; ++e) {
            b0[e] = (_Float16)krow[e];           // d = lh*16+e
            b1[e] = (_Float16)krow[32 + e];      // d = 32+lh*16+e
        }
        v8f c = {};
        c = __builtin_amdgcn_wmma_f32_16x16x32_f16(false, a0, false, b0,
                                                   (short)0, c, false, false);
        c = __builtin_amdgcn_wmma_f32_16x16x32_f16(false, a1, false, b1,
                                                   (short)0, c, false, false);
        // C layout: VGPR r -> M = r + 8*lh ; N = ll
        #pragma unroll
        for (int r = 0; r < 8; ++r)
            sS[m_sub * 16 + r + 8 * lh][n0 + ll] = c[r];
    }
    __syncthreads();

    // ==================== Phase 2: row softmax + attn store =================
    // wave w owns rows 4w..4w+3. exp computed ONCE per element; all LDS/HBM
    // traffic in float4 (global_store_b128 on the 268 MB attn stream).
    float* attn_blk = attn + (size_t)bh * SEQ * SEQ + (size_t)m0 * SEQ;
    for (int rr = 0; rr < 4; ++rr) {
        const int row = w * 4 + rr;
        float* srow = &sS[row][0];
        // pass A: row max
        float mx = -INFINITY;
        for (int i = lane * 4; i < SEQ; i += 128) {
            const float4 x = *(const float4*)(srow + i);
            mx = fmaxf(mx, fmaxf(fmaxf(x.x, x.y), fmaxf(x.z, x.w)));
        }
        #pragma unroll
        for (int off = 16; off; off >>= 1) mx = fmaxf(mx, __shfl_xor(mx, off, 32));
        // pass B: e = exp(s-mx), accumulate sum, keep e in LDS
        float sum = 0.f;
        for (int i = lane * 4; i < SEQ; i += 128) {
            float4 x = *(const float4*)(srow + i);
            x.x = __expf(x.x - mx); x.y = __expf(x.y - mx);
            x.z = __expf(x.z - mx); x.w = __expf(x.w - mx);
            sum += (x.x + x.y) + (x.z + x.w);
            *(float4*)(srow + i) = x;
        }
        #pragma unroll
        for (int off = 16; off; off >>= 1) sum += __shfl_xor(sum, off, 32);
        const float inv = 1.0f / sum;
        // pass C: p = e/sum -> LDS (for P@V) and global attn (once)
        float* arow = attn_blk + (size_t)row * SEQ;
        for (int i = lane * 4; i < SEQ; i += 128) {
            float4 e4 = *(const float4*)(srow + i);
            e4.x *= inv; e4.y *= inv; e4.z *= inv; e4.w *= inv;
            *(float4*)(srow + i) = e4;
            *(float4*)(arow + i) = e4;
        }
    }
    __syncthreads();

    // ======================= Phase 3: O = P @ V =============================
    // wave w: output tile rows 16*(w>>2)..+15, cols 16*(w&3)..+15.
    // V tiles (32x64 f32 = 8 KB) are brought in by the TDM, double-buffered:
    // tile kt+1 streams in while tile kt feeds the WMMAs.
    const int n_sub = w & 3;
    v8f acc = {};

#ifdef USE_TDM
    if (w == 0)
        tdm_issue_v_tile(V, (uint32_t)(uintptr_t)&sV[0][0][0]);
#endif
    for (int kt = 0; kt < 64; ++kt) {
#ifdef USE_TDM
        if (w == 0) {
            if (kt + 1 < 64)   // prefetch next tile into the other buffer
                tdm_issue_v_tile(V + (size_t)(kt + 1) * BM * DIM,
                                 (uint32_t)(uintptr_t)&sV[(kt + 1) & 1][0][0]);
            if (kt == 63) __builtin_amdgcn_s_wait_tensorcnt(0);
            else          __builtin_amdgcn_s_wait_tensorcnt(1);
        }
        __syncthreads();                         // tile kt visible to all waves
#else
        {   // fallback: cooperative VGPR-roundtrip staging of tile kt
            const int kr = tid >> 3, nc = (tid & 7) * 8;
            const float* vp = V + (size_t)(kt * BM + kr) * DIM + nc;
            #pragma unroll
            for (int i = 0; i < 8; ++i) sV[kt & 1][kr][nc + i] = vp[i];
        }
        __syncthreads();
#endif
        // A-fragment from probabilities (f32 LDS -> f16)
        v16h a, b;
        const float* prow = &sS[Mrow][kt * 32 + lh * 16];
        #pragma unroll
        for (int e = 0; e < 16; ++e) {
            a[e]     = (_Float16)prow[e];        // kk = lh*16+e
        }
        // B-fragment: B[kk][n] = V[kt*32+kk][n]; lane: n=n_sub*16+ll, kk=lh*16+e
        const float* vcol = &sV[kt & 1][lh * 16][n_sub * 16 + ll];
        #pragma unroll
        for (int e = 0; e < 16; ++e) b[e] = (_Float16)vcol[e * DIM];

        acc = __builtin_amdgcn_wmma_f32_16x16x32_f16(false, a, false, b,
                                                     (short)0, acc, false, false);
        __syncthreads();                         // tile kt&1 free for reuse
    }

    // ---- store output tile (f32), lanes ll -> consecutive d -> coalesced ----
    float* obase = out + head + (size_t)m0 * DIM;
    #pragma unroll
    for (int r = 0; r < 8; ++r)
        obase[(size_t)(m_sub * 16 + r + 8 * lh) * DIM + n_sub * 16 + ll] = acc[r];
}

extern "C" void kernel_launch(void* const* d_in, const int* in_sizes, int n_in,
                              void* d_out, int out_size, void* d_ws, size_t ws_size,
                              hipStream_t stream)
{
    (void)in_sizes; (void)n_in; (void)d_ws; (void)ws_size; (void)out_size;
    const float* q = (const float*)d_in[0];
    const float* k = (const float*)d_in[1];
    const float* v = (const float*)d_in[2];
    float* out  = (float*)d_out;                       // [2,8,2048,64]
    float* attn = out + (size_t)2 * 8 * 2048 * 64;     // [2,8,2048,2048]

    dim3 grid(SEQ / BM, 2 * 8);   // 64 query blocks x 16 heads
    dim3 block(256);              // 8 wave32
    sdpa_wmma_kernel<<<grid, block, 0, stream>>>(q, k, v, out, attn);
}
